// PrototypeLayer_56968446214767
// MI455X (gfx1250) — matrix operand order repaired and checked by reading, not compiled, not run on hardware
//
#include <hip/hip_runtime.h>

// Problem constants (from reference): B=16, N=256, K=2048, V=768
#define BB 16
#define NN 256
#define KK 2048
#define VV 768
#define MM (BB * NN) // 4096 GEMM rows

typedef __attribute__((ext_vector_type(16))) __bf16 v16bf;
typedef __attribute__((ext_vector_type(8)))  float  v8f;
typedef __attribute__((ext_vector_type(4)))  int    v4i;

union Frag {
  v16bf f;
  v4i   q[2];
};

__device__ __forceinline__ unsigned short f32_to_bf16_rne(float x) {
  unsigned int u = __float_as_uint(x);
  u += 0x7FFFu + ((u >> 16) & 1u);
  return (unsigned short)(u >> 16);
}
__device__ __forceinline__ float bf16_bits_to_f32(unsigned short h) {
  return __uint_as_float(((unsigned int)h) << 16);
}

// ---------------------------------------------------------------------------
// Prologue 1: X (f32, [M, V]) -> Xh/Xl bf16 split + row squared norms
// ---------------------------------------------------------------------------
__global__ __launch_bounds__(256) void prep_x(const float* __restrict__ X,
                                              unsigned short* __restrict__ Xh,
                                              unsigned short* __restrict__ Xl,
                                              float* __restrict__ xsq) {
  const int m = blockIdx.x;
  const int t = threadIdx.x;
  const float* row = X + (size_t)m * VV;
  float s = 0.0f;
  for (int i = t; i < VV; i += 256) {
    float x = row[i];
    unsigned short h = f32_to_bf16_rne(x);
    float xh = bf16_bits_to_f32(h);
    unsigned short l = f32_to_bf16_rne(x - xh);
    Xh[(size_t)m * VV + i] = h;
    Xl[(size_t)m * VV + i] = l;
    s += x * x;
  }
  __shared__ float red[256];
  red[t] = s;
  __syncthreads();
  for (int off = 128; off > 0; off >>= 1) {
    if (t < off) red[t] += red[t + off];
    __syncthreads();
  }
  if (t == 0) xsq[m] = red[0];
}

// ---------------------------------------------------------------------------
// Prologue 2: P (f32, [K, V]) -> Ph/Pl bf16 split + squared norms + tuple copy
// ---------------------------------------------------------------------------
__global__ __launch_bounds__(256) void prep_p(const float* __restrict__ P,
                                              unsigned short* __restrict__ Ph,
                                              unsigned short* __restrict__ Pl,
                                              float* __restrict__ psq,
                                              float* __restrict__ out_protos) {
  const int k = blockIdx.x;
  const int t = threadIdx.x;
  const float* row = P + (size_t)k * VV;
  float s = 0.0f;
  for (int i = t; i < VV; i += 256) {
    float p = row[i];
    unsigned short h = f32_to_bf16_rne(p);
    float ph = bf16_bits_to_f32(h);
    unsigned short l = f32_to_bf16_rne(p - ph);
    Ph[(size_t)k * VV + i] = h;
    Pl[(size_t)k * VV + i] = l;
    out_protos[(size_t)k * VV + i] = p;  // second tuple output: prototypes
    s += p * p;
  }
  __shared__ float red[256];
  red[t] = s;
  __syncthreads();
  for (int off = 128; off > 0; off >>= 1) {
    if (t < off) red[t] += red[t + off];
    __syncthreads();
  }
  if (t == 0) psq[k] = red[0];
}

// ---------------------------------------------------------------------------
// Main GEMM: D[m,n] = xsq[m] + psq[n] - 2 * sum_v X[m,v]*P[n,v]
// bf16x3 split via v_wmma_f32_16x16x32_bf16.
// Block: 128 threads = 4 waves (2x2); block tile 128(M) x 128(N);
// wave tile 64x64 = 4x4 WMMA tiles (16 v8f accumulators).
// ---------------------------------------------------------------------------
__global__ __launch_bounds__(128) void dist_wmma(const unsigned short* __restrict__ Xh,
                                                 const unsigned short* __restrict__ Xl,
                                                 const unsigned short* __restrict__ Ph,
                                                 const unsigned short* __restrict__ Pl,
                                                 const float* __restrict__ xsq,
                                                 const float* __restrict__ psq,
                                                 float* __restrict__ out) {
  const int lane = threadIdx.x & 31;
  const int w    = threadIdx.x >> 5;
  const int wr   = w >> 1;   // 0..1
  const int wc   = w & 1;    // 0..1

  const int m_base = blockIdx.y * 128 + wr * 64;
  const int n_base = blockIdx.x * 128 + wc * 64;

  const int lm      = lane & 15;
  const bool hiHalf = (lane >= 16);

  // A-fragment per-lane element offsets within a 32-wide K chunk
  // (ISA 16-bit A 16x32 layout: lanes 0-15 -> K {0..7,16..23}; lanes 16-31 -> {8..15,24..31})
  const int aoff0 = hiHalf ? 8 : 0;
  const int aoff1 = aoff0 + 16;
  // B-fragment: same K layout as A (lanes 0-15 -> K {0..7,16..23}; lanes 16-31 -> {8..15,24..31})
  const int boff = hiHalf ? 8 : 0;

  // Base row pointers; tile strides (t*16*VV shorts) fold into the 24-bit
  // instruction immediate offset of global_load_b128.
  const unsigned short* aH0 = Xh + (size_t)(m_base + lm) * VV;
  const unsigned short* aL0 = Xl + (size_t)(m_base + lm) * VV;
  const unsigned short* bH0 = Ph + (size_t)(n_base + lm) * VV;
  const unsigned short* bL0 = Pl + (size_t)(n_base + lm) * VV;

  v8f acc[4][4] = {};

  for (int v0 = 0; v0 < VV; v0 += 32) {
    Frag aH[4], aL[4], bH[4], bL[4];
#pragma unroll
    for (int t = 0; t < 4; ++t) {
      const int trow = t * 16 * VV;
      aH[t].q[0] = *(const v4i*)(aH0 + trow + v0 + aoff0);
      aH[t].q[1] = *(const v4i*)(aH0 + trow + v0 + aoff1);
      aL[t].q[0] = *(const v4i*)(aL0 + trow + v0 + aoff0);
      aL[t].q[1] = *(const v4i*)(aL0 + trow + v0 + aoff1);
      bH[t].q[0] = *(const v4i*)(bH0 + trow + v0 + boff);
      bH[t].q[1] = *(const v4i*)(bH0 + trow + v0 + boff + 16);
      bL[t].q[0] = *(const v4i*)(bL0 + trow + v0 + boff);
      bL[t].q[1] = *(const v4i*)(bL0 + trow + v0 + boff + 16);
    }
#pragma unroll
    for (int ti = 0; ti < 4; ++ti) {
#pragma unroll
      for (int tj = 0; tj < 4; ++tj) {
        acc[ti][tj] = __builtin_amdgcn_wmma_f32_16x16x32_bf16(
            false, aH[ti].f, false, bH[tj].f, (short)0, acc[ti][tj], false, false);
        acc[ti][tj] = __builtin_amdgcn_wmma_f32_16x16x32_bf16(
            false, aH[ti].f, false, bL[tj].f, (short)0, acc[ti][tj], false, false);
        acc[ti][tj] = __builtin_amdgcn_wmma_f32_16x16x32_bf16(
            false, aL[ti].f, false, bH[tj].f, (short)0, acc[ti][tj], false, false);
      }
    }
  }

  // Epilogue: C/D layout: VGPR j -> (M = j + (lane>=16 ? 8 : 0), N = lane%16)
#pragma unroll
  for (int ti = 0; ti < 4; ++ti) {
#pragma unroll
    for (int tj = 0; tj < 4; ++tj) {
      const int m0 = m_base + 16 * ti + (hiHalf ? 8 : 0);
      const int n  = n_base + 16 * tj + lm;
      const float pq = psq[n];
#pragma unroll
      for (int j = 0; j < 8; ++j) {
        const int m = m0 + j;
        out[(size_t)m * KK + n] = xsq[m] + pq - 2.0f * acc[ti][tj][j];
      }
    }
  }
}

// ---------------------------------------------------------------------------
// Launch
// ---------------------------------------------------------------------------
extern "C" void kernel_launch(void* const* d_in, const int* in_sizes, int n_in,
                              void* d_out, int out_size, void* d_ws, size_t ws_size,
                              hipStream_t stream) {
  const float* X = (const float*)d_in[0];  // [B,N,V] f32
  const float* P = (const float*)d_in[1];  // [K,V]   f32

  float* out        = (float*)d_out;                   // distances [B,N,K]
  float* out_protos = out + (size_t)MM * KK;           // prototypes copy [K,V]

  // Workspace layout (bytes)
  char* ws = (char*)d_ws;
  const size_t xh_bytes = (size_t)MM * VV * 2;  // 6,291,456
  const size_t ph_bytes = (size_t)KK * VV * 2;  // 3,145,728
  unsigned short* Xh = (unsigned short*)(ws);
  unsigned short* Xl = (unsigned short*)(ws + xh_bytes);
  unsigned short* Ph = (unsigned short*)(ws + 2 * xh_bytes);
  unsigned short* Pl = (unsigned short*)(ws + 2 * xh_bytes + ph_bytes);
  float* xsq = (float*)(ws + 2 * xh_bytes + 2 * ph_bytes);
  float* psq = xsq + MM;

  prep_x<<<MM, 256, 0, stream>>>(X, Xh, Xl, xsq);
  prep_p<<<KK, 256, 0, stream>>>(P, Ph, Pl, psq, out_protos);

  dim3 grid(KK / 128, MM / 128);  // (16, 32)
  dist_wmma<<<grid, 128, 0, stream>>>(Xh, Xl, Ph, Pl, xsq, psq, out);
}
